// AttLayer_14972255993884
// MI455X (gfx1250) — matrix operand order repaired and checked by reading, not compiled, not run on hardware
//
#include <hip/hip_runtime.h>

// Problem constants (from reference): B=16, C=256, C2=128, N=W*H=4096
#define Bsz   16
#define Cch   256
#define C2ch  128
#define Nspat 4096

typedef __attribute__((ext_vector_type(16))) __bf16 v16bf;
typedef __attribute__((ext_vector_type(8)))  __bf16 v8bf;
typedef __attribute__((ext_vector_type(8)))  float  v8f;

// Assemble a 16-element bf16 WMMA fragment from two 16-byte chunks.
__device__ __forceinline__ v16bf ld2x8(const __bf16* p0, const __bf16* p1) {
  v8bf a = *(const v8bf*)p0;
  v8bf b = *(const v8bf*)p1;
  return __builtin_shufflevector(a, b, 0,1,2,3,4,5,6,7,8,9,10,11,12,13,14,15);
}

__device__ __forceinline__ v8f wmma_bf16(v16bf a, v16bf b, v8f c) {
  return __builtin_amdgcn_wmma_f32_16x16x32_bf16(false, a, false, b, (short)0, c,
                                                 false, false);
}

// LDS offset of a __shared__ object: on amdgcn the generic pointer to LDS
// carries the wave-relative LDS byte offset in its low 32 bits.
__device__ __forceinline__ unsigned lds_off(const void* p) {
  return (unsigned)(uintptr_t)p;
}

// Async 16-byte global -> LDS copy (gfx1250, ASYNCcnt-tracked).
__device__ __forceinline__ void async_cp16(unsigned ldsoff, const __bf16* g) {
  asm volatile("global_load_async_to_lds_b128 %0, %1, off"
               :: "v"(ldsoff), "v"(g) : "memory");
}

__device__ __forceinline__ void wait_async0() {
  asm volatile("s_wait_asynccnt 0" ::: "memory");
}
__device__ __forceinline__ void wait_ds0() {
  asm volatile("s_wait_dscnt 0" ::: "memory");
}

// proj[o][n] = sum_c W[o][c] * in[b][c][n] + bias[o], stored as bf16.
// nmajor=1 -> out[(b*N + n)*Cout + o]   (Q, K^T: row = query/key, contiguous channels)
// nmajor=0 -> out[(b*Cout + o)*N + n]   (V: row = channel, contiguous spatial)
__global__ __launch_bounds__(256)
void proj_kernel(const float* __restrict__ in, const float* __restrict__ W,
                 const float* __restrict__ bias, __bf16* __restrict__ out,
                 int Cout, int nmajor)
{
  const int n = blockIdx.x * blockDim.x + threadIdx.x;
  const int o = blockIdx.y;
  const int b = blockIdx.z;
  const float* xp = in + (size_t)(b * Cch) * Nspat + n;
  const float* wp = W + (size_t)o * Cch;
  float acc = bias[o];
#pragma unroll 8
  for (int c = 0; c < Cch; ++c)
    acc = fmaf(wp[c], xp[(size_t)c * Nspat], acc);
  const size_t idx = nmajor ? ((size_t)(b * Nspat + n)) * Cout + o
                            : ((size_t)(b * Cout + o)) * Nspat + n;
  out[idx] = (__bf16)acc;
}

// Flash attention. 256 threads = 8 waves: wave = (qs, ch), qs in 0..3 = 16-query
// strip, ch in 0..1 = 128-channel half. ch==0 waves ("producers") compute
// S = Q*K^T + online softmax once per strip and publish P/alpha through LDS;
// all 8 waves accumulate their own 8 output tiles (16x128 channels).
// K/V tiles are staged into double-buffered LDS with async global->LDS copies.
__global__ __launch_bounds__(256)
void attn_kernel(const __bf16* __restrict__ q,   // (B, N, C2) bf16
                 const __bf16* __restrict__ kT,  // (B, N, C2) bf16
                 const __bf16* __restrict__ v,   // (B, C, N)  bf16
                 const float*  __restrict__ y,   // (B, C, N)  f32
                 const float*  __restrict__ gamma,
                 float* __restrict__ out)        // (B, C, N)  f32
{
  __shared__ __align__(16) __bf16 Kl[2][32][C2ch];   // 16 KB  key tiles
  __shared__ __align__(16) __bf16 Vl[2][Cch][32];    // 32 KB  value tiles
  __shared__ __align__(16) __bf16 Pl[4][16][32];     //  4 KB  P per strip
  __shared__ float Al[4][16];                        // alpha per strip/row
  __shared__ float Ll[4][16];                        // 1/l per strip/row

  const int bb   = blockIdx.y;
  const int qtb  = blockIdx.x * 64;
  const int tid  = threadIdx.x;
  const int wave = tid >> 5;
  const int lane = tid & 31;
  const int qs   = wave & 3;
  const int ch   = wave >> 2;
  const int ln   = lane & 15;
  const int lh   = lane >> 4;
  const int m0   = qtb + qs * 16;

  const __bf16* kTb = kT + (size_t)bb * Nspat * C2ch;
  const __bf16* vb  = v  + (size_t)bb * Cch * Nspat;

  // ---- Q A-fragments (16 rows x 128 K), used by producers ----
  const __bf16* qrow = q + ((size_t)(bb * Nspat + m0 + ln)) * C2ch + 16 * lh;
  v16bf aq[4];
#pragma unroll
  for (int f = 0; f < 4; ++f)
    aq[f] = ld2x8(qrow + f * 32, qrow + f * 32 + 8);

  const v8f vzero = {};
  v8f Ot[8];
#pragma unroll
  for (int t = 0; t < 8; ++t) Ot[t] = vzero;
  v8f mrow, lrow;
#pragma unroll
  for (int i = 0; i < 8; ++i) { mrow[i] = -3.0e38f; lrow[i] = 0.0f; }

  // ---- cooperative async staging of one 32-key K/V tile pair ----
  auto stage = [&](int kt, int bsel) {
    const unsigned kOff = lds_off(&Kl[bsel][0][0]);
    const unsigned vOff = lds_off(&Vl[bsel][0][0]);
    // K tile: 32 rows x 256 B = 512 chunks of 16 B
#pragma unroll
    for (int j = 0; j < 2; ++j) {
      const int cq  = tid + 256 * j;
      const int row = cq >> 4, c16 = cq & 15;
      async_cp16(kOff + row * 256 + c16 * 16,
                 kTb + (size_t)(kt + row) * C2ch + c16 * 8);
    }
    // V tile: 256 rows x 64 B = 1024 chunks of 16 B
#pragma unroll
    for (int j = 0; j < 4; ++j) {
      const int cq  = tid + 256 * j;
      const int row = cq >> 2, c4 = cq & 3;
      async_cp16(vOff + row * 64 + c4 * 16,
                 vb + (size_t)row * Nspat + kt + c4 * 8);
    }
  };

  stage(0, 0);   // prologue: first tile in flight

#pragma unroll 1
  for (int kt = 0; kt < Nspat; kt += 32) {
    const int bsel = (kt >> 5) & 1;
    wait_async0();          // staged tile (bsel) landed in LDS
    wait_ds0();             // all our LDS reads of buffer bsel^1 retired
    __syncthreads();        // whole block: data visible, prev compute done
    if (kt + 32 < Nspat) stage(kt + 32, bsel ^ 1);   // overlap next copy

    if (__builtin_amdgcn_readfirstlane(ch) == 0) {   // scalar branch: EXEC stays full
      // ---- S = Q * K^T over two 16-key column tiles (from LDS) ----
      const __bf16* krow0 = &Kl[bsel][ln][16 * lh];
      const __bf16* krow1 = &Kl[bsel][16 + ln][16 * lh];
      v8f S0 = vzero, S1 = vzero;
#pragma unroll
      for (int f = 0; f < 4; ++f) {
        v16bf b0 = ld2x8(krow0 + f * 32, krow0 + f * 32 + 8);
        v16bf b1 = ld2x8(krow1 + f * 32, krow1 + f * 32 + 8);
        S0 = wmma_bf16(aq[f], b0, S0);
        S1 = wmma_bf16(aq[f], b1, S1);
      }

      // ---- online softmax (reg i <-> row i + 8*lh of the strip) ----
      v8f t;
#pragma unroll
      for (int i = 0; i < 8; ++i) t[i] = fmaxf(S0[i], S1[i]);
#pragma unroll
      for (int xm = 1; xm <= 8; xm <<= 1)
#pragma unroll
        for (int i = 0; i < 8; ++i) t[i] = fmaxf(t[i], __shfl_xor(t[i], xm, 32));

      v8f mnew, alpha;
#pragma unroll
      for (int i = 0; i < 8; ++i) {
        mnew[i]  = fmaxf(mrow[i], t[i]);
        alpha[i] = __expf(mrow[i] - mnew[i]);
        mrow[i]  = mnew[i];
        S0[i] = __expf(S0[i] - mnew[i]);   // S -> P
        S1[i] = __expf(S1[i] - mnew[i]);
      }
      v8f ps = S0 + S1;
#pragma unroll
      for (int xm = 1; xm <= 8; xm <<= 1)
#pragma unroll
        for (int i = 0; i < 8; ++i) ps[i] += __shfl_xor(ps[i], xm, 32);
      lrow = lrow * alpha + ps;

      // ---- publish P (bf16) and alpha through LDS ----
      if (ln == 0) {
#pragma unroll
        for (int i = 0; i < 8; ++i) Al[qs][i + 8 * lh] = alpha[i];
      }
#pragma unroll
      for (int i = 0; i < 8; ++i) {
        Pl[qs][i + 8 * lh][ln]      = (__bf16)S0[i];
        Pl[qs][i + 8 * lh][16 + ln] = (__bf16)S1[i];
      }
      wait_ds0();
    }
    __syncthreads();        // P/alpha ready for both channel halves

    // ---- all waves: rescale accumulators, O += P * V (own 128 channels) ----
    v8f al;
#pragma unroll
    for (int i = 0; i < 8; ++i) al[i] = Al[qs][i + 8 * lh];
#pragma unroll
    for (int t2 = 0; t2 < 8; ++t2) Ot[t2] *= al;

    v16bf pa = ld2x8(&Pl[qs][ln][16 * lh], &Pl[qs][ln][16 * lh + 8]);
#pragma unroll
    for (int ct = 0; ct < 8; ++ct) {
      const __bf16* vrow = &Vl[bsel][ch * 128 + ct * 16 + ln][16 * lh];
      v16bf bv = ld2x8(vrow, vrow + 8);
      Ot[ct] = wmma_bf16(pa, bv, Ot[ct]);
    }
  }

  // ---- epilogue: share 1/l, then out = gamma * O/l + y ----
  if (__builtin_amdgcn_readfirstlane(ch) == 0) {
    if (ln == 0) {
#pragma unroll
      for (int i = 0; i < 8; ++i) Ll[qs][i + 8 * lh] = 1.0f / lrow[i];
    }
    wait_ds0();
  }
  __syncthreads();

  const float g = gamma[0];
#pragma unroll
  for (int ct = 0; ct < 8; ++ct) {
    const int c = ch * 128 + ct * 16 + ln;
#pragma unroll
    for (int i = 0; i < 8; ++i) {
      const float linv = Ll[qs][i + 8 * lh];
      const size_t idx = ((size_t)(bb * Cch + c)) * Nspat + (m0 + i + 8 * lh);
      out[idx] = g * (Ot[ct][i] * linv) + y[idx];
    }
  }
}

extern "C" void kernel_launch(void* const* d_in, const int* in_sizes, int n_in,
                              void* d_out, int out_size, void* d_ws, size_t ws_size,
                              hipStream_t stream) {
  const float* x  = (const float*)d_in[0];
  const float* y  = (const float*)d_in[1];
  const float* Wq = (const float*)d_in[2];
  const float* bq = (const float*)d_in[3];
  const float* Wk = (const float*)d_in[4];
  const float* bk = (const float*)d_in[5];
  const float* Wv = (const float*)d_in[6];
  const float* bv = (const float*)d_in[7];
  const float* gm = (const float*)d_in[8];
  float* out = (float*)d_out;

  // Workspace: Q (16MiB) + K^T (16MiB) + V (32MiB) bf16 scratch.
  __bf16* qb = (__bf16*)d_ws;
  __bf16* kb = qb + (size_t)Bsz * Nspat * C2ch;
  __bf16* vb = kb + (size_t)Bsz * Nspat * C2ch;

  proj_kernel<<<dim3(Nspat / 256, C2ch, Bsz), dim3(256), 0, stream>>>(x, Wq, bq, qb, C2ch, 1);
  proj_kernel<<<dim3(Nspat / 256, C2ch, Bsz), dim3(256), 0, stream>>>(y, Wk, bk, kb, C2ch, 1);
  proj_kernel<<<dim3(Nspat / 256, Cch,  Bsz), dim3(256), 0, stream>>>(y, Wv, bv, vb, Cch, 0);
  attn_kernel<<<dim3(Nspat / 64, Bsz), dim3(256), 0, stream>>>(qb, kb, vb, y, gm, out);
}
